// Gcn_11295763988538
// MI455X (gfx1250) — matrix operand order repaired and checked by reading, not compiled, not run on hardware
//
#include <hip/hip_runtime.h>
#include <math.h>

// ---------------------------------------------------------------------------
// GCN stack on MI455X (gfx1250).
//  - Dense H@W via bf16 WMMA (v_wmma_f32_16x16x32_bf16), 128x64 macro-tile per
//    block: weight slab staged in LDS (ds_load/store_b128), A fragment reused
//    across 4 column tiles (4 WMMAs per A load).
//  - Edge aggregation: float4 gather + global_atomic_add_f32 scatter (the
//    memory-bound part, ~4 GB total -> ~170us at 23.3 TB/s HBM).
// Strides: fp32 H/T rows LDF=320 (>= max padded dout), bf16 K rows LDK=320.
// ---------------------------------------------------------------------------

#define LDF 320
#define LDK 320
#define NFIN 256
#define BN 64           // output columns per block
#define KLDS_MAX 328    // LDS row stride (K + 8 pad), K <= 320

typedef __attribute__((ext_vector_type(16))) __bf16 v16bf;
typedef __attribute__((ext_vector_type(8)))  float  v8f;

union FragU { v16bf v; uint4 u[2]; };

__device__ __forceinline__ unsigned short f2bf(float f) {
    unsigned int u = __float_as_uint(f);
    u += 0x7FFFu + ((u >> 16) & 1u);          // round-to-nearest-even
    return (unsigned short)(u >> 16);
}

// ---- degree / normalization ------------------------------------------------
__global__ void k_fill1(float* p, int n) {
    int i = blockIdx.x * blockDim.x + threadIdx.x;
    if (i < n) p[i] = 1.0f;                    // self-loop contributes 1
}

__global__ void k_deg(const long long* __restrict__ ei, float* deg, int E) {
    int e = blockIdx.x * blockDim.x + threadIdx.x;
    if (e < E) atomicAdd(&deg[(int)ei[(long)E + e]], 1.0f);   // dst row
}

__global__ void k_rsqrt(float* p, int n) {
    int i = blockIdx.x * blockDim.x + threadIdx.x;
    if (i < n) p[i] = rsqrtf(p[i]);            // deg >= 1 always (self-loops)
}

// ---- fp32 -> bf16 conversions ---------------------------------------------
__global__ void k_cvt_h(const float* __restrict__ src, int ldsrc, int kin, int kp32,
                        unsigned short* __restrict__ dst, int n, long total) {
    long tid = (long)blockIdx.x * blockDim.x + threadIdx.x;
    if (tid >= total) return;
    int  k   = (int)(tid % kp32);
    long row = tid / kp32;
    float v = (row < n && k < kin) ? src[row * (long)ldsrc + k] : 0.0f;
    dst[row * LDK + k] = f2bf(v);
}

// W (kin x dout, row-major) -> Wt (dpadg x LDK) transposed, zero padded
__global__ void k_cvt_w(const float* __restrict__ W, int kin, int dout, int kp32,
                        unsigned short* __restrict__ Wt, long total) {
    long tid = (long)blockIdx.x * blockDim.x + threadIdx.x;
    if (tid >= total) return;
    int k  = (int)(tid % kp32);
    int nr = (int)(tid / kp32);
    float v = (nr < dout && k < kin) ? W[(long)k * dout + nr] : 0.0f;
    Wt[(long)nr * LDK + k] = f2bf(v);
}

// ---- WMMA GEMM: block computes 128 rows x 64 cols --------------------------
// 8 waves/block; weight slab (64 cols x K bf16) staged in LDS once per block;
// each wave: 1 A fragment load -> 4 WMMAs against 4 B fragments from LDS.
// Fragment layouts per CDNA5 ISA 7.12.2:
//   A 16x32 bf16: lane row = lane&15, kbase = lane<16?0:8,
//     elems 0..7 -> K=kbase+e, 8..15 -> K=kbase+16+(e-8)   (two 16B chunks)
//   B 32x16 bf16: lane col = lane&15, same K mapping as A (two 16B chunks)
//   C/D: elem v -> row = v + (lane<16?0:8), col = lane&15
__global__ void k_gemm(const unsigned short* __restrict__ A,
                       const unsigned short* __restrict__ B,
                       float* __restrict__ T,
                       int Nblocks, int K) {
    __shared__ unsigned short Bs[BN * KLDS_MAX];
    const int KLDS = K + 8;

    int bm = (int)blockIdx.x / Nblocks;
    int bn = (int)blockIdx.x - bm * Nblocks;

    // cooperative stage of the 64-column weight slab into LDS (b128 chunks)
    const int cpk    = K >> 3;                 // uint4 chunks per column
    const int chunks = BN * cpk;
    for (int idx = threadIdx.x; idx < chunks; idx += blockDim.x) {
        int col = idx / cpk;
        int ch  = idx - col * cpk;
        uint4 v = *(const uint4*)(B + (size_t)(bn * BN + col) * LDK + ch * 8);
        *(uint4*)(&Bs[col * KLDS + ch * 8]) = v;
    }
    __syncthreads();

    const int wave = threadIdx.x >> 5;
    const int lane = threadIdx.x & 31;
    const int l16  = lane & 15;
    const int hi   = lane >> 4;
    const int row0 = (bm * 8 + wave) * 16;

    const unsigned short* arow  = A + (size_t)(row0 + l16) * LDK + hi * 8;
    const unsigned short* bbase = &Bs[(size_t)l16 * KLDS + hi * 8];

    v8f acc0 = {0.f,0.f,0.f,0.f,0.f,0.f,0.f,0.f};
    v8f acc1 = acc0, acc2 = acc0, acc3 = acc0;

    for (int k0 = 0; k0 < K; k0 += 32) {
        FragU a;
        a.u[0] = *(const uint4*)(arow + k0);
        a.u[1] = *(const uint4*)(arow + k0 + 16);
        FragU b0, b1, b2, b3;
        const unsigned short* bp = bbase + k0;
        b0.u[0] = *(const uint4*)(bp);                 b0.u[1] = *(const uint4*)(bp + 16);
        b1.u[0] = *(const uint4*)(bp + 16 * KLDS);     b1.u[1] = *(const uint4*)(bp + 16 * KLDS + 16);
        b2.u[0] = *(const uint4*)(bp + 32 * KLDS);     b2.u[1] = *(const uint4*)(bp + 32 * KLDS + 16);
        b3.u[0] = *(const uint4*)(bp + 48 * KLDS);     b3.u[1] = *(const uint4*)(bp + 48 * KLDS + 16);
        acc0 = __builtin_amdgcn_wmma_f32_16x16x32_bf16(false, a.v, false, b0.v, (short)0, acc0, false, false);
        acc1 = __builtin_amdgcn_wmma_f32_16x16x32_bf16(false, a.v, false, b1.v, (short)0, acc1, false, false);
        acc2 = __builtin_amdgcn_wmma_f32_16x16x32_bf16(false, a.v, false, b2.v, (short)0, acc2, false, false);
        acc3 = __builtin_amdgcn_wmma_f32_16x16x32_bf16(false, a.v, false, b3.v, (short)0, acc3, false, false);
    }

    float* out = T + (size_t)(row0 + hi * 8) * LDF + bn * BN + l16;
#pragma unroll
    for (int v = 0; v < 8; v++) out[(size_t)v * LDF +  0] = acc0[v];
#pragma unroll
    for (int v = 0; v < 8; v++) out[(size_t)v * LDF + 16] = acc1[v];
#pragma unroll
    for (int v = 0; v < 8; v++) out[(size_t)v * LDF + 32] = acc2[v];
#pragma unroll
    for (int v = 0; v < 8; v++) out[(size_t)v * LDF + 48] = acc3[v];
}

// ---- aggregation: H[dst] = sum_e norm(e) * T[src] + selfnorm * T[i] --------
__global__ void k_self(const float* __restrict__ T, const float* __restrict__ dis,
                       float* __restrict__ H, long total, int c4) {
    long tid = (long)blockIdx.x * blockDim.x + threadIdx.x;
    if (tid >= total) return;
    int  cg = (int)(tid % c4);
    long i  = tid / c4;
    float w = dis[i]; w *= w;                  // self-loop norm = 1/deg
    float4 t = *(const float4*)(T + i * LDF + cg * 4);
    float4 r; r.x = t.x * w; r.y = t.y * w; r.z = t.z * w; r.w = t.w * w;
    *(float4*)(H + i * LDF + cg * 4) = r;
}

__global__ void k_edge(const long long* __restrict__ ei, const float* __restrict__ T,
                       const float* __restrict__ dis, float* __restrict__ H,
                       long total, int c4, int E) {
    long tid = (long)blockIdx.x * blockDim.x + threadIdx.x;
    if (tid >= total) return;
    int cg = (int)(tid % c4);
    int e  = (int)(tid / c4);
    int s  = (int)ei[e];
    int d  = (int)ei[(long)E + e];
    float w = dis[s] * dis[d];
    float4 t = *(const float4*)(T + (size_t)s * LDF + cg * 4);
    float* h = H + (size_t)d * LDF + cg * 4;
    atomicAdd(h + 0, t.x * w);
    atomicAdd(h + 1, t.y * w);
    atomicAdd(h + 2, t.z * w);
    atomicAdd(h + 3, t.w * w);
}

__global__ void k_biasrelu(float* __restrict__ H, const float* __restrict__ b,
                           long total, int dout) {
    long tid = (long)blockIdx.x * blockDim.x + threadIdx.x;
    if (tid >= total) return;
    int  c = (int)(tid % dout);
    long i = tid / dout;
    float v = H[i * LDF + c] + b[c];
    H[i * LDF + c] = v > 0.f ? v : 0.f;
}

// ---- final linear + sigmoid ------------------------------------------------
__global__ void k_final(const float* __restrict__ H, const float* __restrict__ wl,
                        const float* __restrict__ bl, float* __restrict__ out,
                        int n, int dh) {
    int i = blockIdx.x * blockDim.x + threadIdx.x;
    if (i >= n) return;
    float s = 0.f;
    for (int c = 0; c < dh; c += 4) {
        float4 h = *(const float4*)(H + (size_t)i * LDF + c);
        float4 w = *(const float4*)(wl + c);
        s += h.x * w.x + h.y * w.y + h.z * w.z + h.w * w.w;
    }
    s += bl[0];
    out[i] = 1.0f / (1.0f + expf(-s));
}

// ---------------------------------------------------------------------------
extern "C" void kernel_launch(void* const* d_in, const int* in_sizes, int n_in,
                              void* d_out, int out_size, void* d_ws, size_t ws_size,
                              hipStream_t stream) {
    const float*     x  = (const float*)d_in[0];
    const long long* ei = (const long long*)d_in[1];
    const int n = in_sizes[0] / NFIN;
    const int E = in_sizes[1] / 2;
    const int npad = (n + 127) & ~127;         // rows padded to 128 (macro-tile)

    char* ws = (char*)d_ws;
    size_t off = 0;
    auto carve = [&](size_t bytes) -> char* {
        char* p = ws + off;
        off = (off + bytes + 255) & ~(size_t)255;
        return p;
    };
    float*          dis  = (float*)carve((size_t)n * 4);
    float*          Hcur = (float*)carve((size_t)npad * LDF * 4);
    float*          T    = (float*)carve((size_t)npad * LDF * 4);
    unsigned short* Hbf  = (unsigned short*)carve((size_t)npad * LDK * 2);
    unsigned short* Wt   = (unsigned short*)carve((size_t)320 * LDK * 2);
    (void)ws_size;

    const int TB = 256;
    auto blocks = [](long total) { return (unsigned)((total + 255) / 256); };

    // symmetric normalization coefficients
    k_fill1<<<blocks(n), TB, 0, stream>>>(dis, n);
    k_deg  <<<blocks(E), TB, 0, stream>>>(ei, dis, E);
    k_rsqrt<<<blocks(n), TB, 0, stream>>>(dis, n);

    struct Layer { const float* W; const float* b; int kin; int dout; };
    Layer L[8];
    for (int l = 0; l < 8; ++l) {
        L[l].W = (const float*)d_in[2 + 2 * l];
        L[l].b = (const float*)d_in[3 + 2 * l];
    }
    L[0].kin = NFIN; L[0].dout = 300;
    L[1].kin = 300;  L[1].dout = 300;
    L[2].kin = 300;  L[2].dout = 300;
    L[3].kin = 300;  L[3].dout = 300;
    L[4].kin = 300;  L[4].dout = 100;
    L[5].kin = 100;  L[5].dout = 100;
    L[6].kin = 100;  L[6].dout = 100;
    L[7].kin = 100;  L[7].dout = 100;

    const float* hsrc = x;
    int ldsrc = NFIN;

    for (int l = 0; l < 8; ++l) {
        const int kin   = L[l].kin;
        const int dout  = L[l].dout;
        const int kp32  = (kin + 31) & ~31;    // K padded for 32-wide WMMA steps
        const int dpadg = (dout + 63) & ~63;   // cols padded to 64 (macro-tile)
        const int c4    = dout / 4;            // 300 and 100 both divisible by 4

        long tc = (long)npad * kp32;
        k_cvt_h<<<blocks(tc), TB, 0, stream>>>(hsrc, ldsrc, kin, kp32, Hbf, n, tc);

        long tw = (long)dpadg * kp32;
        k_cvt_w<<<blocks(tw), TB, 0, stream>>>(L[l].W, kin, dout, kp32, Wt, tw);

        const int Mblocks = npad / 128;
        const int Nblocks = dpadg / BN;
        k_gemm<<<(unsigned)(Mblocks * Nblocks), TB, 0, stream>>>(Hbf, Wt, T, Nblocks, kp32);

        long ts = (long)n * c4;
        k_self<<<blocks(ts), TB, 0, stream>>>(T, dis, Hcur, ts, c4);

        long te = (long)E * c4;
        k_edge<<<blocks(te), TB, 0, stream>>>(ei, T, dis, Hcur, te, c4, E);

        long tb = (long)n * dout;
        k_biasrelu<<<blocks(tb), TB, 0, stream>>>(Hcur, L[l].b, tb, dout);

        hsrc = Hcur; ldsrc = LDF;
    }

    k_final<<<blocks(n), TB, 0, stream>>>(Hcur, (const float*)d_in[18],
                                          (const float*)d_in[19], (float*)d_out, n, 100);
}
